// DeepSeekBlock_14139032338631
// MI455X (gfx1250) — matrix operand order repaired and checked by reading, not compiled, not run on hardware
//
#include <hip/hip_runtime.h>
#include <math.h>

// ---------------------------------------------------------------------------
// DeepSeek-style block (MLA attention + MoE) for gfx1250 (MI455X).
// Strategy: one-time fp32->bf16 conversion of weights/activations feeding
// GEMMs; all GEMMs use v_wmma_f32_16x16x32_bf16 with bf16 operands staged
// into double-buffered LDS via CDNA5 async copies
// (global_load_async_to_lds_b128 + s_wait_asynccnt).
// ---------------------------------------------------------------------------

typedef __attribute__((ext_vector_type(16))) __bf16 v16bf;
typedef __attribute__((ext_vector_type(8)))  __bf16 v8bf;
typedef __attribute__((ext_vector_type(4)))  __bf16 v4bf;
typedef __attribute__((ext_vector_type(8)))  float  v8f;

constexpr int kH  = 1024;   // hidden
constexpr int kNH = 16;     // heads
constexpr int kHD = 64;     // head dim
constexpr int kRD = 32;     // rope dim
constexpr int kL  = 256;    // latent
constexpr int kER = 7;      // routed experts
constexpr int kI  = 2048;   // ffn inner
constexpr int kB  = 4;
constexpr int kT  = 1024;
constexpr int kN  = kB * kT;      // 4096 tokens
constexpr float kEPS = 1e-5f;

__device__ __forceinline__ __bf16 f2bf(float f) {
  union { float f; unsigned u; } v; v.f = f;
  unsigned r = (v.u + 0x7FFFu + ((v.u >> 16) & 1u)) >> 16;  // RNE
  unsigned short h = (unsigned short)r;
  __bf16 b; __builtin_memcpy(&b, &h, sizeof(b));
  return b;
}

// ---------------------------------------------------------------------------
// CDNA5 async copy global->LDS (ASYNCcnt-tracked, no VGPR staging).
// ---------------------------------------------------------------------------
#define USE_ASYNC_COPY 1

#if USE_ASYNC_COPY
typedef __attribute__((address_space(3))) const void lds_cvoid_t;
__device__ __forceinline__ unsigned lds_off_of(const void* p) {
  return (unsigned)(unsigned long long)(lds_cvoid_t*)p;
}
__device__ __forceinline__ void async_copy_b128(unsigned lds_dst, const void* gsrc) {
  asm volatile("global_load_async_to_lds_b128 %0, %1, off"
               :: "v"(lds_dst), "v"((unsigned long long)gsrc)
               : "memory");
}
__device__ __forceinline__ void wait_async_all() {
  asm volatile("s_wait_asynccnt 0x0" ::: "memory");
}
#endif

// ---------------------------------------------------------------------------
// WMMA GEMM: R[M,N] = A[M,K] * B[N,K]^T, bf16 operands, fp32 accumulate.
// mode 0: Cf = R ; 1: Cf += R ; 2: Cf += rowscale[row*rs]*R ; 3: Cb = bf16(R)
// Requires M%128==0, N%128==0, K%64==0 (true for every GEMM here).
// Block tile 128x128x64, 8 waves (2x4), wave tile 64x32 (4x2 WMMA tiles),
// double-buffered dynamic LDS (73728 B), 16 v_wmma per k-iteration.
// ---------------------------------------------------------------------------
constexpr int GBM = 128, GBN = 128, GBK = 64;
constexpr int GLDK = GBK + 8;     // 144B row stride: 16B aligned, 64-bank clean
constexpr int kGemmLds = 2 * 2 * GBM * GLDK * (int)sizeof(__bf16);  // 73728

__global__ __launch_bounds__(256)
void gemm_bf16_wmma(const __bf16* __restrict__ A, const __bf16* __restrict__ Bm,
                    float* __restrict__ Cf, __bf16* __restrict__ Cb,
                    const float* __restrict__ rowscale,
                    int M, int N, int K, int mode, int rs_stride) {
  extern __shared__ __bf16 smem[];
  __bf16 (*As)[GLDK] = (__bf16(*)[GLDK])smem;                     // [2*GBM][GLDK]
  __bf16 (*Bs)[GLDK] = (__bf16(*)[GLDK])(smem + 2 * GBM * GLDK);  // [2*GBN][GLDK]

  const int tid  = threadIdx.x;
  const int lane = tid & 31;
  const int wave = tid >> 5;
  const int bm = blockIdx.y * GBM;
  const int bn = blockIdx.x * GBN;
  const int wrow = (wave & 1) * 64;
  const int wcol = (wave >> 1) * 32;

  v8f acc[4][2];
#pragma unroll
  for (int mt = 0; mt < 4; ++mt)
#pragma unroll
    for (int nt = 0; nt < 2; ++nt) { v8f z = {}; acc[mt][nt] = z; }

  // staging map: 256 threads x 4 passes x 16B cover a 128x64 bf16 tile
  const int lr = tid >> 3;          // 0..31
  const int lc = (tid & 7) * 8;     // elem col: 0,8,...,56

  const int ktiles = K / GBK;

  auto issue_tile = [&](int kt, int buf) {
    const size_t ka = (size_t)kt * GBK;
#pragma unroll
    for (int p = 0; p < 4; ++p) {
      const int r = lr + p * 32;
      const __bf16* ga = &A [(size_t)(bm + r) * K + ka + lc];
      const __bf16* gb = &Bm[(size_t)(bn + r) * K + ka + lc];
#if USE_ASYNC_COPY
      async_copy_b128(lds_off_of(&As[buf * GBM + r][lc]), ga);
      async_copy_b128(lds_off_of(&Bs[buf * GBN + r][lc]), gb);
#else
      *(uint4*)&As[buf * GBM + r][lc] = *(const uint4*)ga;
      *(uint4*)&Bs[buf * GBN + r][lc] = *(const uint4*)gb;
#endif
    }
  };

  // ISA frag maps (wave32):
  // A 16x32: lane L -> row (L&15); 8 contig K at kbase & kbase+16, kbase=(L>=16)*8
  // B 32x16: lane L -> col (L&15); 8 contig K at kbase & kbase+16, kbase=(L>=16)*8
  const int ar  = wrow + (lane & 15);
  const int akb = (lane >> 4) << 3;
  const int bc  = wcol + (lane & 15);
  const int bkb = (lane >> 4) << 3;

  issue_tile(0, 0);
  for (int kt = 0; kt < ktiles; ++kt) {
    const int buf = kt & 1;
#if USE_ASYNC_COPY
    wait_async_all();
#endif
    __syncthreads();
    if (kt + 1 < ktiles) issue_tile(kt + 1, buf ^ 1);

#pragma unroll
    for (int kk = 0; kk < 2; ++kk) {   // two 32-deep WMMA steps per 64-K slab
      const int kb = kk * 32;
      v16bf af[4], bfr[2];
#pragma unroll
      for (int mt = 0; mt < 4; ++mt) {
        const v8bf lo = *(const v8bf*)(&As[buf * GBM + ar + mt * 16][kb + akb]);
        const v8bf hi = *(const v8bf*)(&As[buf * GBM + ar + mt * 16][kb + 16 + akb]);
#pragma unroll
        for (int i = 0; i < 8; ++i) { af[mt][i] = lo[i]; af[mt][8 + i] = hi[i]; }
      }
#pragma unroll
      for (int nt = 0; nt < 2; ++nt) {
        const v8bf lo = *(const v8bf*)(&Bs[buf * GBN + bc + nt * 16][kb + bkb]);
        const v8bf hi = *(const v8bf*)(&Bs[buf * GBN + bc + nt * 16][kb + bkb + 16]);
#pragma unroll
        for (int i = 0; i < 8; ++i) { bfr[nt][i] = lo[i]; bfr[nt][8 + i] = hi[i]; }
      }
#pragma unroll
      for (int mt = 0; mt < 4; ++mt)
#pragma unroll
        for (int nt = 0; nt < 2; ++nt)
          acc[mt][nt] = __builtin_amdgcn_wmma_f32_16x16x32_bf16(
              false, af[mt], false, bfr[nt], (short)0, acc[mt][nt], false, false);
    }
  }

  // Epilogue. C/D layout: VGPR r -> M = r + 8*(lane>=16), N = lane&15.
  const int moff = (lane >> 4) << 3;
#pragma unroll
  for (int mt = 0; mt < 4; ++mt)
#pragma unroll
    for (int nt = 0; nt < 2; ++nt) {
      const int col = bn + wcol + nt * 16 + (lane & 15);
#pragma unroll
      for (int r = 0; r < 8; ++r) {
        const int row = bm + wrow + mt * 16 + moff + r;
        const size_t idx = (size_t)row * N + col;
        const float val = acc[mt][nt][r];
        if (mode == 0)      Cf[idx] = val;
        else if (mode == 1) Cf[idx] += val;
        else if (mode == 2) Cf[idx] += rowscale[(size_t)row * rs_stride] * val;
        else                Cb[idx] = f2bf(val);
      }
    }
}

// ---------------------------------------------------------------------------
// fp32 -> bf16 bulk conversion (vectorized; n % 4 == 0 for all tensors here)
// ---------------------------------------------------------------------------
__global__ void cvt_bf16_kernel(const float* __restrict__ s, __bf16* __restrict__ d,
                                size_t n4) {
  const size_t i = (size_t)blockIdx.x * blockDim.x + threadIdx.x;
  if (i < n4) {
    const float4 v = ((const float4*)s)[i];
    v4bf o; o[0] = f2bf(v.x); o[1] = f2bf(v.y); o[2] = f2bf(v.z); o[3] = f2bf(v.w);
    *(v4bf*)&d[i * 4] = o;
  }
}

// ---------------------------------------------------------------------------
// LayerNorm: one 256-thread block per token, bf16 output (feeds GEMMs).
// ---------------------------------------------------------------------------
__global__ __launch_bounds__(256)
void ln_kernel(const float* __restrict__ x, const float* __restrict__ w,
               __bf16* __restrict__ out) {
  __shared__ float red[256];
  const int n = blockIdx.x, tid = threadIdx.x;
  const float* xr = x + (size_t)n * kH;
  float s = 0.f;
  for (int j = tid; j < kH; j += 256) s += xr[j];
  red[tid] = s; __syncthreads();
  for (int o = 128; o > 0; o >>= 1) { if (tid < o) red[tid] += red[tid + o]; __syncthreads(); }
  const float mu = red[0] * (1.f / kH);
  __syncthreads();
  float v = 0.f;
  for (int j = tid; j < kH; j += 256) { const float d = xr[j] - mu; v += d * d; }
  red[tid] = v; __syncthreads();
  for (int o = 128; o > 0; o >>= 1) { if (tid < o) red[tid] += red[tid + o]; __syncthreads(); }
  const float inv = rsqrtf(red[0] * (1.f / kH) + kEPS);
  for (int j = tid; j < kH; j += 256)
    out[(size_t)n * kH + j] = f2bf((xr[j] - mu) * inv * w[j]);
}

// ---------------------------------------------------------------------------
// RoPE in place on [N, NH, RD] fp32 buffers.
// ---------------------------------------------------------------------------
__global__ __launch_bounds__(256)
void rope_kernel(float* __restrict__ qr, float* __restrict__ kr) {
  const size_t gid = (size_t)blockIdx.x * blockDim.x + threadIdx.x;
  if (gid >= (size_t)kN * kNH * (kRD / 2)) return;
  const int i = (int)(gid & 15);
  const size_t nh = gid >> 4;
  const int t = (int)((nh / kNH) % kT);
  const float invf = powf(10000.f, -(2.f * i) / (float)kRD);
  const float ang = (float)t * invf;
  const float c = cosf(ang), s = sinf(ang);
  const size_t base = nh * kRD;
  {
    const float x1 = qr[base + i], x2 = qr[base + i + 16];
    qr[base + i]      = x1 * c - x2 * s;
    qr[base + i + 16] = x2 * c + x1 * s;
  }
  {
    const float x1 = kr[base + i], x2 = kr[base + i + 16];
    kr[base + i]      = x1 * c - x2 * s;
    kr[base + i + 16] = x2 * c + x1 * s;
  }
}

// ---------------------------------------------------------------------------
// Causal attention, streaming softmax; one wave per query row; bf16 output.
// ---------------------------------------------------------------------------
__global__ __launch_bounds__(256)
void attn_kernel(const float* __restrict__ q, const float* __restrict__ k,
                 const float* __restrict__ v, const float* __restrict__ qr,
                 const float* __restrict__ kr, __bf16* __restrict__ y) {
  const int wave = threadIdx.x >> 5, lane = threadIdx.x & 31;
  const int t = blockIdx.x * 8 + wave;
  const int h = blockIdx.y, b = blockIdx.z;
  const size_t n = (size_t)b * kT + t;
  const float q_r = qr[(n * kNH + h) * kRD + lane];
  const float q_n = q [(n * kNH + h) * kHD + 32 + lane];
  float m = -INFINITY, l = 0.f, acc0 = 0.f, acc1 = 0.f;
  for (int j = 0; j <= t; ++j) {
    const size_t nk = (size_t)b * kT + j;
    float s = q_r * kr[(nk * kNH + h) * kRD + lane]
            + q_n * k [(nk * kNH + h) * kHD + 32 + lane];
#pragma unroll
    for (int off = 16; off > 0; off >>= 1) s += __shfl_xor(s, off, 32);
    s *= 0.125f;  // 1/sqrt(64)
    const float mn = fmaxf(m, s);
    const float corr = __expf(m - mn);
    const float p = __expf(s - mn);
    l = l * corr + p;
    acc0 = acc0 * corr + p * v[(nk * kNH + h) * kHD + lane];
    acc1 = acc1 * corr + p * v[(nk * kNH + h) * kHD + 32 + lane];
    m = mn;
  }
  const float il = 1.f / l;
  y[(n * kNH + h) * kHD + lane]      = f2bf(acc0 * il);
  y[(n * kNH + h) * kHD + 32 + lane] = f2bf(acc1 * il);
}

// ---------------------------------------------------------------------------
// Router on bf16 activations: sigmoid -> top-2 -> renormalize.
// ---------------------------------------------------------------------------
__global__ __launch_bounds__(256)
void router_kernel(const __bf16* __restrict__ x, const float* __restrict__ rw,
                   const float* __restrict__ bias, float* __restrict__ wgt) {
  __shared__ float red[kER][256];
  const int n = blockIdx.x, tid = threadIdx.x;
  float p[kER];
#pragma unroll
  for (int e = 0; e < kER; ++e) p[e] = 0.f;
  for (int j = tid; j < kH; j += 256) {
    const float xv = (float)x[(size_t)n * kH + j];
#pragma unroll
    for (int e = 0; e < kER; ++e) p[e] += xv * rw[(size_t)e * kH + j];
  }
#pragma unroll
  for (int e = 0; e < kER; ++e) red[e][tid] = p[e];
  __syncthreads();
  for (int o = 128; o > 0; o >>= 1) {
    if (tid < o)
#pragma unroll
      for (int e = 0; e < kER; ++e) red[e][tid] += red[e][tid + o];
    __syncthreads();
  }
  if (tid == 0) {
    float pr[kER];
#pragma unroll
    for (int e = 0; e < kER; ++e)
      pr[e] = 1.f / (1.f + __expf(-(red[e][0] + bias[e])));
    int i0 = 0;
#pragma unroll
    for (int e = 1; e < kER; ++e) if (pr[e] > pr[i0]) i0 = e;
    int i1 = (i0 == 0) ? 1 : 0;
#pragma unroll
    for (int e = 0; e < kER; ++e) if (e != i0 && pr[e] > pr[i1]) i1 = e;
    const float inv = 1.f / (pr[i0] + pr[i1]);
#pragma unroll
    for (int e = 0; e < kER; ++e) wgt[(size_t)n * kER + e] = 0.f;
    wgt[(size_t)n * kER + i0] = pr[i0] * inv;
    wgt[(size_t)n * kER + i1] = pr[i1] * inv;
  }
}

// ---------------------------------------------------------------------------
// Elementwise helpers.
// ---------------------------------------------------------------------------
__global__ void zero_kernel(float* __restrict__ p, size_t n) {
  const size_t i = (size_t)blockIdx.x * blockDim.x + threadIdx.x;
  if (i < n) p[i] = 0.f;
}
__global__ void add_kernel(const float* __restrict__ a, const float* __restrict__ b,
                           float* __restrict__ o, size_t n) {
  const size_t i = (size_t)blockIdx.x * blockDim.x + threadIdx.x;
  if (i < n) o[i] = a[i] + b[i];
}
__global__ void silu_mul_bf16_kernel(const __bf16* __restrict__ g,
                                     const __bf16* __restrict__ u,
                                     __bf16* __restrict__ act, size_t n) {
  const size_t i = (size_t)blockIdx.x * blockDim.x + threadIdx.x;
  if (i < n) {
    const float x = (float)g[i];
    act[i] = f2bf((x / (1.f + __expf(-x))) * (float)u[i]);
  }
}

// ---------------------------------------------------------------------------
// Host orchestration.
// ---------------------------------------------------------------------------
static inline void gemm(hipStream_t s, const __bf16* A, const __bf16* Bm,
                        float* Cf, __bf16* Cb, int M, int N, int K, int mode,
                        const float* rowscale = nullptr, int rs_stride = 0) {
  dim3 grid(N / GBN, M / GBM);
  gemm_bf16_wmma<<<grid, 256, kGemmLds, s>>>(A, Bm, Cf, Cb, rowscale,
                                             M, N, K, mode, rs_stride);
}

extern "C" void kernel_launch(void* const* d_in, const int* in_sizes, int n_in,
                              void* d_out, int out_size, void* d_ws, size_t ws_size,
                              hipStream_t stream) {
  hipFuncSetAttribute((const void*)gemm_bf16_wmma,
                      hipFuncAttributeMaxDynamicSharedMemorySize, kGemmLds);

  const float* x        = (const float*)d_in[0];
  const float* ln1_w    = (const float*)d_in[1];
  const float* ln2_w    = (const float*)d_in[2];
  const float* router_w = (const float*)d_in[11];
  const float* r_bias   = (const float*)d_in[12];
  float* out = (float*)d_out;

  // ---- workspace ----
  char* base = (char*)d_ws;
  size_t off = 0;
  auto allocf = [&](size_t n) { float* p = (float*)(base + off); off += ((n * 4 + 15) & ~15ull); return p; };
  auto allocb = [&](size_t n) { __bf16* p = (__bf16*)(base + off); off += ((n * 2 + 15) & ~15ull); return p; };

  float* qbuf = allocf((size_t)kN * kH);
  float* kbuf = allocf((size_t)kN * kH);
  float* vbuf = allocf((size_t)kN * kH);
  float* qr   = allocf((size_t)kN * kNH * kRD);
  float* kr   = allocf((size_t)kN * kNH * kRD);
  float* x1   = allocf((size_t)kN * kH);
  float* moe  = allocf((size_t)kN * kH);
  float* wgt  = allocf((size_t)kN * kER);

  __bf16* xn_b  = allocb((size_t)kN * kH);
  __bf16* kvl_b = allocb((size_t)kN * kL);
  __bf16* ql_b  = allocb((size_t)kN * kL);
  __bf16* y_b   = allocb((size_t)kN * kH);
  __bf16* h2_b  = allocb((size_t)kN * kH);
  __bf16* g_b   = allocb((size_t)kN * kI);
  __bf16* u_b   = allocb((size_t)kN * kI);
  __bf16* act_b = allocb((size_t)kN * kI);

  // bf16 weight copies (converted every call; deterministic)
  const size_t wLH = (size_t)kL * kH, wHL = (size_t)kNH * kHD * kL;
  const size_t wIH = (size_t)kI * kH;
  __bf16* kv_d_b   = allocb(wLH);
  __bf16* q_d_b    = allocb(wLH);
  __bf16* k_u_b    = allocb(wHL);
  __bf16* q_u_b    = allocb(wHL);
  __bf16* v_u_b    = allocb(wHL);
  __bf16* rope_q_b = allocb((size_t)kNH * kRD * kL);
  __bf16* rope_k_b = allocb((size_t)kNH * kRD * kH);
  __bf16* o_w_b    = allocb((size_t)kH * kH);
  __bf16* sh_g_b   = allocb(wIH);
  __bf16* sh_u_b   = allocb(wIH);
  __bf16* sh_d_b   = allocb(wIH);
  __bf16* rt_g_b   = allocb((size_t)kER * wIH);
  __bf16* rt_u_b   = allocb((size_t)kER * wIH);
  __bf16* rt_d_b   = allocb((size_t)kER * wIH);
  (void)ws_size; (void)in_sizes; (void)n_in; (void)out_size;

  const int EB = 256;
  auto eg = [](size_t n) { return (unsigned)((n + 255) / 256); };
  auto cvt = [&](const void* src, __bf16* dst, size_t n) {
    cvt_bf16_kernel<<<eg(n / 4), EB, 0, stream>>>((const float*)src, dst, n / 4);
  };

  cvt(d_in[3],  kv_d_b,   wLH);
  cvt(d_in[4],  q_d_b,    wLH);
  cvt(d_in[5],  k_u_b,    wHL);
  cvt(d_in[6],  q_u_b,    wHL);
  cvt(d_in[7],  v_u_b,    (size_t)kNH * kHD * kL);
  cvt(d_in[8],  rope_q_b, (size_t)kNH * kRD * kL);
  cvt(d_in[9],  rope_k_b, (size_t)kNH * kRD * kH);
  cvt(d_in[10], o_w_b,    (size_t)kH * kH);
  cvt(d_in[13], sh_g_b,   wIH);
  cvt(d_in[14], sh_u_b,   wIH);
  cvt(d_in[15], sh_d_b,   wIH);
  cvt(d_in[16], rt_g_b,   (size_t)kER * wIH);
  cvt(d_in[17], rt_u_b,   (size_t)kER * wIH);
  cvt(d_in[18], rt_d_b,   (size_t)kER * wIH);

  const size_t NH_ = (size_t)kN * kH;
  const size_t NI_ = (size_t)kN * kI;

  // ---- attention ----
  ln_kernel<<<kN, 256, 0, stream>>>(x, ln1_w, xn_b);
  gemm(stream, xn_b,  kv_d_b,   nullptr, kvl_b, kN, kL,        kH, 3);
  gemm(stream, xn_b,  q_d_b,    nullptr, ql_b,  kN, kL,        kH, 3);
  gemm(stream, kvl_b, k_u_b,    kbuf, nullptr,  kN, kNH * kHD, kL, 0);
  gemm(stream, ql_b,  q_u_b,    qbuf, nullptr,  kN, kNH * kHD, kL, 0);
  gemm(stream, kvl_b, v_u_b,    vbuf, nullptr,  kN, kNH * kHD, kL, 0);
  gemm(stream, ql_b,  rope_q_b, qr,   nullptr,  kN, kNH * kRD, kL, 0);
  gemm(stream, xn_b,  rope_k_b, kr,   nullptr,  kN, kNH * kRD, kH, 0);
  rope_kernel<<<eg((size_t)kN * kNH * (kRD / 2)), EB, 0, stream>>>(qr, kr);
  attn_kernel<<<dim3(kT / 8, kNH, kB), 256, 0, stream>>>(qbuf, kbuf, vbuf, qr, kr, y_b);
  hipMemcpyAsync(x1, x, NH_ * sizeof(float), hipMemcpyDeviceToDevice, stream);
  gemm(stream, y_b, o_w_b, x1, nullptr, kN, kH, kH, 1);   // x1 = x + y @ o_w^T

  // ---- MoE ----
  ln_kernel<<<kN, 256, 0, stream>>>(x1, ln2_w, h2_b);
  router_kernel<<<kN, 256, 0, stream>>>(h2_b, router_w, r_bias, wgt);
  zero_kernel<<<eg(NH_), EB, 0, stream>>>(moe, NH_);

  // shared expert (NS == 1)
  gemm(stream, h2_b, sh_g_b, nullptr, g_b, kN, kI, kH, 3);
  gemm(stream, h2_b, sh_u_b, nullptr, u_b, kN, kI, kH, 3);
  silu_mul_bf16_kernel<<<eg(NI_), EB, 0, stream>>>(g_b, u_b, act_b, NI_);
  gemm(stream, act_b, sh_d_b, moe, nullptr, kN, kH, kI, 1);

  // routed experts: dense compute, per-token top-2 weighted combine
  for (int e = 0; e < kER; ++e) {
    gemm(stream, h2_b, rt_g_b + (size_t)e * wIH, nullptr, g_b, kN, kI, kH, 3);
    gemm(stream, h2_b, rt_u_b + (size_t)e * wIH, nullptr, u_b, kN, kI, kH, 3);
    silu_mul_bf16_kernel<<<eg(NI_), EB, 0, stream>>>(g_b, u_b, act_b, NI_);
    gemm(stream, act_b, rt_d_b + (size_t)e * wIH, moe, nullptr, kN, kH, kI, 2,
         wgt + e, kER);
  }

  // out = x1 + moe
  add_kernel<<<eg(NH_), EB, 0, stream>>>(x1, moe, out, NH_);
}
